// LoRIExpertBank_91018946937089
// MI455X (gfx1250) — matrix operand order repaired and checked by reading, not compiled, not run on hardware
//
#include <hip/hip_runtime.h>
#include <hip/hip_bf16.h>

typedef __attribute__((ext_vector_type(16))) _Float16 v16h;
typedef __attribute__((ext_vector_type(8)))  _Float16 v8h;
typedef __attribute__((ext_vector_type(8)))  float    v8f;
typedef __attribute__((ext_vector_type(4)))  float    v4f;

#define IN_F   4096
#define RANK_  64
#define NEXP   8
#define NROWS  16384   // B*S = 4*4096
#define TM     128     // rows per block
#define TD     128     // cols per block
#define TK     32      // K per step
#define KSTEPS 16      // 512 / 32

// Stage one 128x32 X tile (rw*p) and one 128x32 W tile (A*mask), f32 -> f16.
__device__ __forceinline__ void stage_tiles(
    int ks, int m0, int d0, int lrow, int lseg,
    const float* __restrict__ P, const float* __restrict__ RW,
    const float* __restrict__ A, const float* __restrict__ MSK,
    _Float16* __restrict__ xbuf, _Float16* __restrict__ wbuf)
{
  const int e     = ks >> 1;          // expert
  const int rbase = (ks & 1) * 32;    // rank offset within expert

  // ---- X = rw * p ----
  {
    const int gm = m0 + lrow;
    const float w = RW[gm * NEXP + e];
    const float* src = P + gm * RANK_ + rbase + lseg * 16;
    _Float16* dst = xbuf + lrow * TK + lseg * 16;
#pragma unroll
    for (int c = 0; c < 4; ++c) {
      v4f v = *(const v4f*)(src + 4 * c);
#pragma unroll
      for (int j = 0; j < 4; ++j)
        dst[4 * c + j] = (_Float16)(w * v[j]);
    }
  }
  // ---- W = A * mask, stored [d][k] so B frags are K-contiguous per column ----
  {
    const int gd = d0 + lrow;
    const int base = (e * IN_F + gd) * RANK_ + rbase + lseg * 16;
    const float* srcA = A   + base;
    const float* srcM = MSK + base;
    _Float16* dst = wbuf + lrow * TK + lseg * 16;
#pragma unroll
    for (int c = 0; c < 4; ++c) {
      v4f a = *(const v4f*)(srcA + 4 * c);
      v4f m = *(const v4f*)(srcM + 4 * c);
#pragma unroll
      for (int j = 0; j < 4; ++j)
        dst[4 * c + j] = (_Float16)(a[j] * m[j]);
    }
    if (ks + 2 < KSTEPS) {   // warm L2 two steps ahead
      const int en = (ks + 2) >> 1;
      const int rn = ((ks + 2) & 1) * 32;
      const int nbase = (en * IN_F + gd) * RANK_ + rn + lseg * 16;
      __builtin_prefetch(A + nbase, 0, 1);
      __builtin_prefetch(MSK + nbase, 0, 1);
    }
  }
}

__global__ __launch_bounds__(256, 2)
void lori_wmma_kernel(const float* __restrict__ P,    // [NROWS, 64]
                      const float* __restrict__ RW,   // [NROWS, 8]
                      const float* __restrict__ A,    // [8, 4096, 64]
                      const float* __restrict__ MSK,  // [8, 4096, 64]
                      float* __restrict__ OUT)        // [NROWS, 4096]
{
  // 64 KB shared block:
  //  - during the K loop:  sX ping-pong (2*8KB) + sW ping-pong (2*8KB) = 32 KB
  //  - in the epilogue:    sC = 128x128 f32 tile = 64 KB (lifetimes disjoint)
  __shared__ __align__(128) unsigned char smem[64 * 1024];
  _Float16* sX = (_Float16*)smem;                   // [2][TM*TK]
  _Float16* sW = (_Float16*)(smem + 2 * TM * TK * 2); // [2][TD*TK]
  float*    sC = (float*)smem;                      // [TM*TD]

  const int tid   = threadIdx.x;
  const int lane  = tid & 31;
  const int wave  = tid >> 5;
  const int row16 = lane & 15;   // matrix row (A) / col (B,C) within 16
  const int khalf = lane >> 4;   // lane-half: K swizzle select / M half of C

  const int d0 = blockIdx.x * TD;
  const int m0 = blockIdx.y * TM;

  // 8 waves -> 4 (M) x 2 (D): each wave owns a 32x64 strip = 2x4 WMMA tiles
  const int wm = wave & 3;
  const int wd = wave >> 2;

  const int lrow = tid >> 1;   // 0..127: row (X) / column (W) staged by this thread
  const int lseg = tid & 1;    // which 16-element half of the 32-wide K slab

  v8f acc[2][4];
#pragma unroll
  for (int i = 0; i < 2; ++i)
#pragma unroll
    for (int j = 0; j < 4; ++j)
      acc[i][j] = v8f{};

  // Prologue: stage K-step 0 into buffer 0.
  stage_tiles(0, m0, d0, lrow, lseg, P, RW, A, MSK, sX, sW);

  for (int ks = 0; ks < KSTEPS; ++ks) {
    __syncthreads();   // staged data for step ks is visible; previous reads done

    const int cur = ks & 1;
    // Stage ks+1 into the other buffer while computing ks.
    if (ks + 1 < KSTEPS)
      stage_tiles(ks + 1, m0, d0, lrow, lseg, P, RW, A, MSK,
                  sX + (cur ^ 1) * (TM * TK), sW + (cur ^ 1) * (TD * TK));

    const _Float16* xb = sX + cur * (TM * TK);
    const _Float16* wb = sW + cur * (TD * TK);

    // A 16x32 f16 frag, this kernel's contiguous K order (B matches): lane-half 0 holds K 0..15, half 1 K 16..31
    v16h afrag[2];
#pragma unroll
    for (int i = 0; i < 2; ++i) {
      const _Float16* ap = xb + (wm * 32 + i * 16 + row16) * TK + khalf * 16;
      v8h lo = *(const v8h*)(ap);        // K = khalf*16 .. +7
      v8h hi = *(const v8h*)(ap + 8);    // K = khalf*16+8 .. +7
      afrag[i] = __builtin_shufflevector(lo, hi,
                   0,1,2,3,4,5,6,7,8,9,10,11,12,13,14,15);
    }
#pragma unroll
    for (int j = 0; j < 4; ++j) {
      // B 32x16 f16 frag, same contiguous K order as A: lane-half 0 holds K 0..15 of its column, half 1 K 16..31
      const _Float16* bp = wb + (wd * 64 + j * 16 + row16) * TK + khalf * 16;
      v16h bfrag = *(const v16h*)(bp);
#pragma unroll
      for (int i = 0; i < 2; ++i) {
        acc[i][j] = __builtin_amdgcn_wmma_f32_16x16x32_f16(
            false, afrag[i], false, bfrag, (short)0, acc[i][j], false, false);
      }
    }
  }

  // ---- epilogue: C fragments -> LDS (row-major) -> coalesced b128 stores ----
  __syncthreads();   // all waves done reading sX/sW; safe to alias as sC
#pragma unroll
  for (int i = 0; i < 2; ++i) {
#pragma unroll
    for (int j = 0; j < 4; ++j) {
      // C layout: lanes 0-15 hold M = v (N = lane); lanes 16-31 hold M = 8+v
      const int r = wm * 32 + i * 16 + khalf * 8;
      const int c = wd * 64 + j * 16 + row16;
#pragma unroll
      for (int v = 0; v < 8; ++v)
        sC[(r + v) * TD + c] = acc[i][j][v];
    }
  }
  __syncthreads();

  // 128x128 f32 tile = 4096 float4; 16 per thread; each wave-iteration writes
  // one full 512 B row segment -> global_store_b128, perfectly coalesced.
#pragma unroll
  for (int it = 0; it < 16; ++it) {
    const int idx  = it * 256 + tid;   // 0..4095
    const int row  = idx >> 5;         // 32 float4 per 128-col row
    const int col4 = idx & 31;
    v4f val = *(const v4f*)(sC + row * TD + col4 * 4);
    *(v4f*)(OUT + (size_t)(m0 + row) * IN_F + d0 + col4 * 4) = val;
  }
}

extern "C" void kernel_launch(void* const* d_in, const int* in_sizes, int n_in,
                              void* d_out, int out_size, void* d_ws, size_t ws_size,
                              hipStream_t stream) {
  const float* P   = (const float*)d_in[0];  // projected_input [4,4096,64]
  const float* RW  = (const float*)d_in[1];  // routing_weights [4,4096,8]
  const float* A   = (const float*)d_in[2];  // A [8,4096,64]
  const float* MSK = (const float*)d_in[3];  // sparse_mask [8,4096,64]
  float* OUT = (float*)d_out;                // [4,4096,4096]

  dim3 grid(IN_F / TD, NROWS / TM);          // 32 x 128 blocks
  lori_wmma_kernel<<<grid, 256, 0, stream>>>(P, RW, A, MSK, OUT);
}
